// DfNetStreaming_55765855371915
// MI455X (gfx1250) — matrix operand order repaired and checked by reading, not compiled, not run on hardware
//
#include <hip/hip_runtime.h>
#include <hip/hip_bf16.h>
#include <math.h>

// ---------------------------------------------------------------------------
// DfNet streaming step for MI455X (gfx1250, wave32, WMMA).
// All dense math (GRU gate GEMMs, grouped linears, pointwise convs) goes
// through one WMMA f16->f32 GEMM kernel (v_wmma_f32_16x16x32_f16) with
// vectorized b128 staging and register-pipelined global loads.
// ---------------------------------------------------------------------------

typedef _Float16 h16;
typedef __attribute__((ext_vector_type(4)))  float    v4f;
typedef __attribute__((ext_vector_type(8)))  float    v8f;
typedef __attribute__((ext_vector_type(8)))  _Float16 v8h;
typedef __attribute__((ext_vector_type(16))) _Float16 v16h;

#define B_      2048
#define CH_     64
#define NBERB_  32
#define NBDF_   96
#define H_      512
#define OUTCH_  10   // ORDER*2
#define G_      8

__device__ __forceinline__ float actf(float v, int act) {
  switch (act) {
    case 1:  return v > 0.f ? v : 0.f;                 // relu
    case 2:  return 1.f / (1.f + __expf(-v));          // sigmoid
    case 3:  return tanhf(v);                          // tanh
    default: return v;
  }
}

// ---------------------------------------------------------------------------
// WMMA GEMM:  C[g] = act( A[g](f32->f16) @ Wt[g]^T + bias[g] )
// Wt is stored K-transposed per group: Wt[n][k], row stride K.
// Block tile 64x64, K-tile 64 (2 wmma-K chunks). 8 waves: 4(M) x 2(N),
// each wave owns a 16x32 output slab -> 4 v_wmma per K-tile.
// Staging: each thread moves 16 A floats (4x b128 load, cvt, 2x b128 LDS
// store) and 16 W halves (2x b128) per tile; next tile's loads are issued
// before the current tile's WMMAs to hide HBM latency.
// ---------------------------------------------------------------------------
__global__ __launch_bounds__(256) void k_gemm(
    const float* __restrict__ A, long lda, long sAg,
    const h16*  __restrict__ Wt, long sWg,
    const float* __restrict__ bias, long sBg,
    float* __restrict__ C, long ldc, long sCg,
    int M, int N, int K, int act)
{
  __shared__ h16 As[64][72];   // [m][k], row = 144B (16B aligned)
  __shared__ h16 Ws[64][72];   // [n][k]

  const int g = blockIdx.z;
  A  += (long)g * sAg;
  Wt += (long)g * sWg;
  C  += (long)g * sCg;
  const float* bp = bias ? bias + (long)g * sBg : nullptr;

  const int tid  = threadIdx.x;
  const int lane = tid & 31;
  const int wave = tid >> 5;
  const int wm = wave & 3;          // M sub-tile
  const int wn = wave >> 2;         // N 32-wide slab
  const int r  = lane & 15;
  const int hh = lane >> 4;

  const long tileM = (long)blockIdx.y * 64;
  const long tileN = (long)blockIdx.x * 64;

  const int row = tid >> 2;         // 0..63 (staging row)
  const int seg = (tid & 3) * 16;   // staging k-offset

  v8f acc0 = {0.f,0.f,0.f,0.f,0.f,0.f,0.f,0.f};
  v8f acc1 = acc0;

  v4f ar[4];
  v8h wr[2];
  const v4f zf = {0.f, 0.f, 0.f, 0.f};
  const v8h zh = {(h16)0.f, (h16)0.f, (h16)0.f, (h16)0.f,
                  (h16)0.f, (h16)0.f, (h16)0.f, (h16)0.f};

  const long gmRow = tileM + row;
  const long gnRow = tileN + row;
  const bool mOK = gmRow < M;
  const bool nOK = gnRow < N;

  auto ldA = [&](int k0) {
    if (mOK) {
      const v4f* p = (const v4f*)(A + gmRow * lda + k0 + seg);
      ar[0] = p[0]; ar[1] = p[1]; ar[2] = p[2]; ar[3] = p[3];
    } else {
      ar[0] = zf; ar[1] = zf; ar[2] = zf; ar[3] = zf;
    }
  };
  auto ldW = [&](int k0) {
    if (nOK) {
      const v8h* p = (const v8h*)(Wt + gnRow * K + k0 + seg);
      wr[0] = p[0]; wr[1] = p[1];
    } else {
      wr[0] = zh; wr[1] = zh;
    }
  };
  auto stage = [&]() {
    v8h h0, h1;
    h0[0]=(h16)ar[0][0]; h0[1]=(h16)ar[0][1]; h0[2]=(h16)ar[0][2]; h0[3]=(h16)ar[0][3];
    h0[4]=(h16)ar[1][0]; h0[5]=(h16)ar[1][1]; h0[6]=(h16)ar[1][2]; h0[7]=(h16)ar[1][3];
    h1[0]=(h16)ar[2][0]; h1[1]=(h16)ar[2][1]; h1[2]=(h16)ar[2][2]; h1[3]=(h16)ar[2][3];
    h1[4]=(h16)ar[3][0]; h1[5]=(h16)ar[3][1]; h1[6]=(h16)ar[3][2]; h1[7]=(h16)ar[3][3];
    *(v8h*)&As[row][seg]     = h0;
    *(v8h*)&As[row][seg + 8] = h1;
    *(v8h*)&Ws[row][seg]     = wr[0];
    *(v8h*)&Ws[row][seg + 8] = wr[1];
  };

  ldA(0);
  ldW(0);
  for (int k0 = 0; k0 < K; k0 += 64) {
    stage();
    __syncthreads();
    if (k0 + 64 < K) {      // issue next tile's global loads early
      ldA(k0 + 64);
      ldW(k0 + 64);
    }
#pragma unroll
    for (int kc = 0; kc < 2; ++kc) {
      const int kb = kc * 32;
      // A fragment: row wm*16+r, contiguous K half {kb+hh*16..+15}
      const h16* pa = &As[wm * 16 + r][kb];
      v8h alo = *(const v8h*)(pa + hh * 16);
      v8h ahi = *(const v8h*)(pa + hh * 16 + 8);
      v16h af = __builtin_shufflevector(alo, ahi,
                  0,1,2,3,4,5,6,7,8,9,10,11,12,13,14,15);
      // B fragments: column n, contiguous K half {kb+hh*16..+15}
      const h16* pb0 = &Ws[wn * 32 + r][kb];
      v8h b0l = *(const v8h*)(pb0 + hh * 16);
      v8h b0h = *(const v8h*)(pb0 + hh * 16 + 8);
      v16h bf0 = __builtin_shufflevector(b0l, b0h,
                  0,1,2,3,4,5,6,7,8,9,10,11,12,13,14,15);
      const h16* pb1 = &Ws[wn * 32 + 16 + r][kb];
      v8h b1l = *(const v8h*)(pb1 + hh * 16);
      v8h b1h = *(const v8h*)(pb1 + hh * 16 + 8);
      v16h bf1 = __builtin_shufflevector(b1l, b1h,
                  0,1,2,3,4,5,6,7,8,9,10,11,12,13,14,15);
      acc0 = __builtin_amdgcn_wmma_f32_16x16x32_f16(false, af, false, bf0,
                                                    (short)0, acc0, false, false);
      acc1 = __builtin_amdgcn_wmma_f32_16x16x32_f16(false, af, false, bf1,
                                                    (short)0, acc1, false, false);
    }
    __syncthreads();
  }

  // epilogue: C layout -> row = v + 8*hh, col = r
  const long col0 = tileN + wn * 32 + r;
  const long col1 = col0 + 16;
#pragma unroll
  for (int v = 0; v < 8; ++v) {
    const long orow = tileM + wm * 16 + v + 8 * hh;
    if (orow >= M) continue;
    if (col0 < N) {
      float o = acc0[v] + (bp ? bp[col0] : 0.f);
      C[orow * ldc + col0] = actf(o, act);
    }
    if (col1 < N) {
      float o = acc1[v] + (bp ? bp[col1] : 0.f);
      C[orow * ldc + col1] = actf(o, act);
    }
  }
}

// ---------------------------------------------------------------------------
// Weight conversion / folding (write K-transposed f16: dst[g][n][k])
// ---------------------------------------------------------------------------
__global__ void k_cvt_t(const float* __restrict__ s, h16* __restrict__ d,
                        int K, int N, long total) {
  long i = (long)blockIdx.x * blockDim.x + threadIdx.x;
  if (i >= total) return;
  const long gkn = (long)K * N;
  const long g = i / gkn;
  const long rem = i - g * gkn;
  const int k = (int)(rem / N), n = (int)(rem % N);
  d[g * gkn + (long)n * K + k] = (h16)s[i];
}

// dt[co*C + c] = pw[c*Co+co] * scale[co] * (dw ? dw[c] : 1)
__global__ void k_fold_pw_t(const float* __restrict__ pw, const float* __restrict__ dw,
                            const float* __restrict__ scale, h16* __restrict__ d,
                            int C, int Co) {
  int i = blockIdx.x * blockDim.x + threadIdx.x;
  if (i >= C * Co) return;
  const int co = i % Co, c = i / Co;
  float v = pw[i] * scale[co];
  if (dw) v *= dw[c];
  d[(long)co * C + c] = (h16)v;
}

// ---------------------------------------------------------------------------
// Elementwise helpers
// ---------------------------------------------------------------------------
__global__ void k_add(const float* __restrict__ a, const float* __restrict__ b,
                      float* __restrict__ o, long n) {
  long i = (long)blockIdx.x * blockDim.x + threadIdx.x;
  if (i < n) o[i] = a[i] + b[i];
}

// GRU gate fuse (keras reset_after, gate order z,r,n). Writes two outputs
// (scratch + strided slice of d_out state region).
__global__ void k_gru_gate(const float* __restrict__ gx, const float* __restrict__ gh,
                           const float* __restrict__ h, long hStride, long hOff,
                           float* __restrict__ out1, long o1Stride, long o1Off,
                           float* __restrict__ out2, long o2Stride, long o2Off,
                           int Bn, int Hd) {
  long i = (long)blockIdx.x * blockDim.x + threadIdx.x;
  if (i >= (long)Bn * Hd) return;
  const int b = (int)(i / Hd), j = (int)(i % Hd);
  const long gb = (long)b * 3 * Hd;
  const float xz = gx[gb + j],          hz = gh[gb + j];
  const float xr = gx[gb + Hd + j],     hr = gh[gb + Hd + j];
  const float xn = gx[gb + 2 * Hd + j], hn = gh[gb + 2 * Hd + j];
  const float z  = 1.f / (1.f + __expf(-(xz + hz)));
  const float rr = 1.f / (1.f + __expf(-(xr + hr)));
  const float nn = tanhf(xn + rr * hn);
  const float hv = h[(long)b * hStride + hOff + j];
  const float o  = z * hv + (1.f - z) * nn;
  out1[(long)b * o1Stride + o1Off + j] = o;
  if (out2) out2[(long)b * o2Stride + o2Off + j] = o;
}

// dense [B,K] @ [K,1]: out = sigmoid(dot + b)*scale + off   (K % 4 == 0)
__global__ void k_dense1(const float* __restrict__ x, const float* __restrict__ W,
                         const float* __restrict__ bb, float* __restrict__ out,
                         int K, float scale, float off) {
  int b = blockIdx.x * blockDim.x + threadIdx.x;
  if (b >= B_) return;
  float acc = 0.f;
  const v4f* xv = (const v4f*)(x + (long)b * K);
  const v4f* wv = (const v4f*)W;
  for (int k = 0; k < (K >> 2); ++k) {
    const v4f a = xv[k], w = wv[k];
    acc += a[0]*w[0] + a[1]*w[1] + a[2]*w[2] + a[3]*w[3];
  }
  out[b] = (1.f / (1.f + __expf(-(acc + bb[0])))) * scale + off;
}

// ---------------------------------------------------------------------------
// Depthwise kernels
// ---------------------------------------------------------------------------
__global__ void k_dw_freq3(const float* __restrict__ x, const float* __restrict__ xadd,
                           const float* __restrict__ dw, float* __restrict__ y,
                           int Fin, int Fout, int stride) {
  long i = (long)blockIdx.x * blockDim.x + threadIdx.x;
  if (i >= (long)B_ * Fout * CH_) return;
  const int c = (int)(i % CH_);
  const int f = (int)((i / CH_) % Fout);
  const int b = (int)(i / ((long)CH_ * Fout));
  float s = 0.f;
  for (int kf = 0; kf < 3; ++kf) {
    const int fi = f * stride + kf - 1;
    if (fi < 0 || fi >= Fin) continue;
    const long idx = ((long)b * Fin + fi) * CH_ + c;
    float v = x[idx];
    if (xadd) v += xadd[idx];
    s += dw[kf * CH_ + c] * v;
  }
  y[i] = s;
}

// transpose freq conv depthwise: lhs_dilation 2, pad (1,2), flipped (1,3) kernel
__global__ void k_dwt_freq3(const float* __restrict__ x, const float* __restrict__ xadd,
                            const float* __restrict__ dw, float* __restrict__ y,
                            int Fin) {
  const int Fout = 2 * Fin;
  long i = (long)blockIdx.x * blockDim.x + threadIdx.x;
  if (i >= (long)B_ * Fout * CH_) return;
  const int c  = (int)(i % CH_);
  const int fo = (int)((i / CH_) % Fout);
  const int b  = (int)(i / ((long)CH_ * Fout));
  float s = 0.f;
  for (int k = 0; k < 3; ++k) {
    const int q = fo - 1 + k;               // dilated coordinate
    if (q < 0 || (q & 1)) continue;
    const int fi = q >> 1;
    if (fi >= Fin) continue;
    const long idx = ((long)b * Fin + fi) * CH_ + c;
    float v = x[idx];
    if (xadd) v += xadd[idx];
    s += dw[(2 - k) * CH_ + c] * v;         // flipped kernel
  }
  y[i] = s;
}

// erb_conv0: C_in=1, 3x3 time-freq with 2-frame state buffer; fused pointwise
// 1->64 + BN + relu, and new_erb_buf update.
__global__ void k_erb_conv0(const float* __restrict__ frame,  // [B,32]
                            const float* __restrict__ buf,    // [B,2,32]
                            const float* __restrict__ dw,     // [3*3]
                            const float* __restrict__ pw,     // [64]
                            const float* __restrict__ scale, const float* __restrict__ bias,
                            float* __restrict__ e0,           // [B,32,64]
                            float* __restrict__ newbuf) {     // [B,2,32]
  long i = (long)blockIdx.x * blockDim.x + threadIdx.x;
  if (i >= (long)B_ * NBERB_) return;
  const int f = (int)(i % NBERB_), b = (int)(i / NBERB_);
  float s = 0.f;
  for (int kt = 0; kt < 3; ++kt)
    for (int kf = 0; kf < 3; ++kf) {
      const int fi = f + kf - 1;
      if (fi < 0 || fi >= NBERB_) continue;
      const float v = (kt < 2) ? buf[((long)b * 2 + kt) * NBERB_ + fi]
                               : frame[(long)b * NBERB_ + fi];
      s += dw[kt * 3 + kf] * v;
    }
  float* eo = e0 + i * CH_;
  for (int co = 0; co < CH_; ++co) {
    const float o = s * pw[co] * scale[co] + bias[co];
    eo[co] = o > 0.f ? o : 0.f;
  }
  newbuf[((long)b * 2 + 0) * NBERB_ + f] = buf[((long)b * 2 + 1) * NBERB_ + f];
  newbuf[((long)b * 2 + 1) * NBERB_ + f] = frame[i];
}

// df_conv0: C_in=2 (complex), 3x3 with 2-frame buffer; fused pw 2->64 + relu,
// and new_df_buf update.
__global__ void k_df_conv0(const float* __restrict__ spec,    // feat_spec [B,2,96]
                           const float* __restrict__ buf,     // [B,2,96,2]
                           const float* __restrict__ dw,      // [(kt*3+kf)*2 + c]
                           const float* __restrict__ pw,      // [2][64]
                           const float* __restrict__ scale, const float* __restrict__ bias,
                           float* __restrict__ c0,            // [B,96,64]
                           float* __restrict__ newbuf) {      // [B,2,96,2]
  long i = (long)blockIdx.x * blockDim.x + threadIdx.x;
  if (i >= (long)B_ * NBDF_) return;
  const int f = (int)(i % NBDF_), b = (int)(i / NBDF_);
  float s0 = 0.f, s1 = 0.f;
  for (int kt = 0; kt < 3; ++kt)
    for (int kf = 0; kf < 3; ++kf) {
      const int fi = f + kf - 1;
      if (fi < 0 || fi >= NBDF_) continue;
      float v0, v1;
      if (kt < 2) {
        const long base = ((long)b * 2 + kt) * NBDF_ + fi;
        v0 = buf[base * 2 + 0]; v1 = buf[base * 2 + 1];
      } else {
        v0 = spec[((long)b * 2 + 0) * NBDF_ + fi];
        v1 = spec[((long)b * 2 + 1) * NBDF_ + fi];
      }
      s0 += dw[(kt * 3 + kf) * 2 + 0] * v0;
      s1 += dw[(kt * 3 + kf) * 2 + 1] * v1;
    }
  float* co0 = c0 + i * CH_;
  for (int co = 0; co < CH_; ++co) {
    const float o = (s0 * pw[co] + s1 * pw[CH_ + co]) * scale[co] + bias[co];
    co0[co] = o > 0.f ? o : 0.f;
  }
  for (int c = 0; c < 2; ++c) {
    newbuf[(((long)b * 2 + 0) * NBDF_ + f) * 2 + c] =
        buf[(((long)b * 2 + 1) * NBDF_ + f) * 2 + c];
    newbuf[(((long)b * 2 + 1) * NBDF_ + f) * 2 + c] =
        spec[((long)b * 2 + c) * NBDF_ + f];
  }
}

// df_convp depthwise: time-5 kernel over [4-frame buffer, c0]; also emits
// new_convp_buf (shift + append c0).
__global__ void k_dw_time5(const float* __restrict__ c0,   // [B,96,64]
                           const float* __restrict__ buf,  // [B,4,96,64]
                           const float* __restrict__ dw,   // [kt*64+c]
                           float* __restrict__ y,          // [B*96,64]
                           float* __restrict__ newbuf) {   // [B,4,96,64]
  long i = (long)blockIdx.x * blockDim.x + threadIdx.x;
  if (i >= (long)B_ * NBDF_ * CH_) return;
  const int c = (int)(i % CH_);
  const int f = (int)((i / CH_) % NBDF_);
  const int b = (int)(i / ((long)CH_ * NBDF_));
  const float xc = c0[i];
  float s = dw[4 * CH_ + c] * xc;
  for (int kt = 0; kt < 4; ++kt)
    s += dw[kt * CH_ + c] * buf[(((long)b * 4 + kt) * NBDF_ + f) * CH_ + c];
  y[i] = s;
  newbuf[(((long)b * 4 + 3) * NBDF_ + f) * CH_ + c] = xc;
  for (int kt = 0; kt < 3; ++kt)
    newbuf[(((long)b * 4 + kt) * NBDF_ + f) * CH_ + c] =
        buf[(((long)b * 4 + kt + 1) * NBDF_ + f) * CH_ + c];
}

// conv0_out: (1,3) depthwise + pw 64->1 + BN + sigmoid over (p0 + x1); writes m.
__global__ void k_conv0_out(const float* __restrict__ p0, const float* __restrict__ x1,
                            const float* __restrict__ dw,   // [kf*64+c]
                            const float* __restrict__ pw,   // [64]
                            const float* __restrict__ scale, const float* __restrict__ bias,
                            float* __restrict__ m) {        // [B,32]
  long i = (long)blockIdx.x * blockDim.x + threadIdx.x;
  if (i >= (long)B_ * NBERB_) return;
  const int f = (int)(i % NBERB_), b = (int)(i / NBERB_);
  float acc = 0.f;
  for (int c = 0; c < CH_; ++c) {
    float t = 0.f;
    for (int kf = 0; kf < 3; ++kf) {
      const int fi = f + kf - 1;
      if (fi < 0 || fi >= NBERB_) continue;
      const long idx = ((long)b * NBERB_ + fi) * CH_ + c;
      t += dw[kf * CH_ + c] * (p0[idx] + x1[idx]);
    }
    acc += t * pw[c];
  }
  const float v = acc * scale[0] + bias[0];
  m[i] = 1.f / (1.f + __expf(-v));
}

// coefs assembly: coefs[b,ord,0,f,ri] = tanh_glin[b, f*10+oc] + c0f[b,f,oc]
__global__ void k_coefs(const float* __restrict__ gout,  // [B,960] (tanh applied)
                        const float* __restrict__ c0f,   // [B,96,10]
                        float* __restrict__ coefs) {     // [B,5,96,2]
  long i = (long)blockIdx.x * blockDim.x + threadIdx.x;
  if (i >= (long)B_ * NBDF_ * OUTCH_) return;
  const int oc = (int)(i % OUTCH_);
  const int f  = (int)((i / OUTCH_) % NBDF_);
  const int b  = (int)(i / ((long)OUTCH_ * NBDF_));
  const float v = gout[i] + c0f[i];
  const int ord = oc >> 1, ri = oc & 1;
  coefs[(((long)b * 5 + ord) * NBDF_ + f) * 2 + ri] = v;
}

// ---------------------------------------------------------------------------
// Host orchestration
// ---------------------------------------------------------------------------
static inline int nb(long n) { return (int)((n + 255) / 256); }

extern "C" void kernel_launch(void* const* d_in, const int* in_sizes, int n_in,
                              void* d_out, int out_size, void* d_ws, size_t ws_size,
                              hipStream_t stream) {
  (void)in_sizes; (void)n_in; (void)out_size; (void)ws_size;

  const float* feat_erb   = (const float*)d_in[0];   // [B,1,1,32]
  const float* feat_spec  = (const float*)d_in[1];   // [B,2,1,96]
  const float* enc_state  = (const float*)d_in[2];   // [B,512]
  const float* erb_state  = (const float*)d_in[3];   // [B,512]
  const float* df_state   = (const float*)d_in[4];   // [B,1024]
  const float* erb0_buf   = (const float*)d_in[5];   // [B,2,32,1]
  const float* df0_buf    = (const float*)d_in[6];   // [B,2,96,2]
  const float* convp_buf  = (const float*)d_in[7];   // [B,4,96,64]
  auto P = [&](int i) { return (const float*)d_in[i]; };
  enum {
    I_c0out = 8, I_c0p = 12, I_c1p = 16, I_c2p = 20, I_c3p = 24,
    I_ct1 = 28, I_ct2 = 32, I_ct3 = 36, I_dfc0 = 40, I_dfc1 = 44, I_dfcp = 48,
    I_dfa_W = 52, I_dfa_b = 53, I_fcemb = 54,
    I_df0_Wh = 55, I_df0_Wx = 56, I_df0_bh = 57, I_df0_bx = 58,
    I_df1_Wh = 59, I_df1_Wx = 60, I_df1_bh = 61, I_df1_bx = 62,
    I_df_win = 63, I_dfout = 64,
    I_enc_Wh = 65, I_enc_Wx = 66, I_enc_bh = 67, I_enc_bx = 68, I_enc_win = 69,
    I_ec0 = 70, I_ec1 = 74, I_ec2 = 78, I_ec3 = 82,
    I_erb_Wh = 86, I_erb_Wx = 87, I_erb_bh = 88, I_erb_bx = 89,
    I_erb_win = 90, I_erb_wout = 91,
    I_lsnr_W = 92, I_lsnr_b = 93
  };

  float* out = (float*)d_out;
  const long O_m      = 0;
  const long O_coefs  = O_m + (long)B_ * NBERB_;
  const long O_alpha  = O_coefs + (long)B_ * 5 * NBDF_ * 2;
  const long O_lsnr   = O_alpha + B_;
  const long O_enc    = O_lsnr + B_;
  const long O_erbst  = O_enc + (long)B_ * H_;
  const long O_dfst   = O_erbst + (long)B_ * H_;
  const long O_erbbuf = O_dfst + (long)B_ * 2 * H_;
  const long O_dfbuf  = O_erbbuf + (long)B_ * 2 * NBERB_;
  const long O_cpbuf  = O_dfbuf + (long)B_ * 2 * NBDF_ * 2;

  size_t off = 0;
  auto alloc = [&](size_t bytes) -> void* {
    void* p = (char*)d_ws + off;
    off += (bytes + 255) & ~(size_t)255;
    return p;
  };
  auto af32 = [&](long n) { return (float*)alloc((size_t)n * 4); };
  auto af16 = [&](long n) { return (h16*)alloc((size_t)n * 2); };

  // f16 weights (K-transposed)
  h16 *Wx_enc = af16(512 * 1536), *Wh_enc = af16(512 * 1536);
  h16 *Wx_erb = af16(512 * 1536), *Wh_erb = af16(512 * 1536);
  h16 *Wx_df0 = af16(512 * 1536), *Wh_df0 = af16(512 * 1536);
  h16 *Wx_df1 = af16(512 * 1536), *Wh_df1 = af16(512 * 1536);
  h16 *Gw_enc = af16(8 * 64 * 64),  *Gw_fce = af16(8 * 384 * 64);
  h16 *Gw_ein = af16(8 * 64 * 64),  *Gw_eout = af16(8 * 64 * 64);
  h16 *Gw_din = af16(8 * 64 * 64),  *Gw_dout = af16(8 * 64 * 120);
  h16 *Pw_e1 = af16(64 * 64), *Pw_e2 = af16(64 * 64), *Pw_e3 = af16(64 * 64);
  h16 *Pw_d1 = af16(64 * 64);
  h16 *Pw_c3p = af16(64 * 64), *Pw_t3 = af16(64 * 64);
  h16 *Pw_c2p = af16(64 * 64), *Pw_t2 = af16(64 * 64);
  h16 *Pw_c1p = af16(64 * 64), *Pw_t1 = af16(64 * 64);
  h16 *Pw_c0p = af16(64 * 64), *Pw_cp = af16(64 * 10);

  // f32 activations
  float *e0 = af32((long)B_ * 32 * 64), *e1 = af32((long)B_ * 16 * 64);
  float *e2 = af32((long)B_ * 8 * 64),  *e3 = af32((long)B_ * 8 * 64);
  float *c0 = af32((long)B_ * 96 * 64), *c1 = af32((long)B_ * 48 * 64);
  float *cemb = af32((long)B_ * 512),   *emb = af32((long)B_ * 512);
  float *embO = af32((long)B_ * 512),   *xin = af32((long)B_ * 512);
  float *ta = af32((long)B_ * 512), *tb = af32((long)B_ * 512);
  float *tc = af32((long)B_ * 512), *td = af32((long)B_ * 512);
  float *gx = af32((long)B_ * 1536), *gh = af32((long)B_ * 1536);
  float *abuf = af32((long)B_ * 96 * 64);
  float *px = af32((long)B_ * 32 * 64), *xx = af32((long)B_ * 32 * 64);
  float *c0f = af32((long)B_ * 96 * 10), *gout = af32((long)B_ * 960);

  auto gemm = [&](const float* A, long lda, long sAg,
                  const h16* Wt, long sWg,
                  const float* bias, long sBg,
                  float* C, long ldc, long sCg,
                  int M, int N, int K, int act, int groups) {
    dim3 grid((N + 63) / 64, (M + 63) / 64, groups);
    k_gemm<<<grid, 256, 0, stream>>>(A, lda, sAg, Wt, sWg, bias, sBg,
                                     C, ldc, sCg, M, N, K, act);
  };
  auto cvtT = [&](int idx, h16* dst, int K, int N, int groups) {
    k_cvt_t<<<nb((long)groups * K * N), 256, 0, stream>>>(
        P(idx), dst, K, N, (long)groups * K * N);
  };
  auto fold = [&](int pwI, int dwI, int scI, h16* dst, int C, int Co) {
    k_fold_pw_t<<<nb((long)C * Co), 256, 0, stream>>>(
        P(pwI), dwI >= 0 ? P(dwI) : nullptr, P(scI), dst, C, Co);
  };

  // ================= Phase 0: weight prep (K-transposed f16) =================
  cvtT(I_enc_Wx, Wx_enc, 512, 1536, 1); cvtT(I_enc_Wh, Wh_enc, 512, 1536, 1);
  cvtT(I_erb_Wx, Wx_erb, 512, 1536, 1); cvtT(I_erb_Wh, Wh_erb, 512, 1536, 1);
  cvtT(I_df0_Wx, Wx_df0, 512, 1536, 1); cvtT(I_df0_Wh, Wh_df0, 512, 1536, 1);
  cvtT(I_df1_Wx, Wx_df1, 512, 1536, 1); cvtT(I_df1_Wh, Wh_df1, 512, 1536, 1);
  cvtT(I_enc_win, Gw_enc, 64, 64, 8);
  cvtT(I_fcemb,   Gw_fce, 384, 64, 8);
  cvtT(I_erb_win, Gw_ein, 64, 64, 8);
  cvtT(I_erb_wout, Gw_eout, 64, 64, 8);
  cvtT(I_df_win,  Gw_din, 64, 64, 8);
  cvtT(I_dfout,   Gw_dout, 64, 120, 8);
  fold(I_ec1 + 2, -1, I_ec1 + 3, Pw_e1, 64, 64);
  fold(I_ec2 + 2, -1, I_ec2 + 3, Pw_e2, 64, 64);
  fold(I_ec3 + 2, -1, I_ec3 + 3, Pw_e3, 64, 64);
  fold(I_dfc1 + 2, -1, I_dfc1 + 3, Pw_d1, 64, 64);
  fold(I_c3p + 2, I_c3p + 1, I_c3p + 3, Pw_c3p, 64, 64);
  fold(I_ct3 + 2, -1, I_ct3 + 3, Pw_t3, 64, 64);
  fold(I_c2p + 2, I_c2p + 1, I_c2p + 3, Pw_c2p, 64, 64);
  fold(I_ct2 + 2, -1, I_ct2 + 3, Pw_t2, 64, 64);
  fold(I_c1p + 2, I_c1p + 1, I_c1p + 3, Pw_c1p, 64, 64);
  fold(I_ct1 + 2, -1, I_ct1 + 3, Pw_t1, 64, 64);
  fold(I_c0p + 2, I_c0p + 1, I_c0p + 3, Pw_c0p, 64, 64);
  fold(I_dfcp + 2, -1, I_dfcp + 3, Pw_cp, 64, 10);

  // ================= Phase 1: encoder =================
  k_erb_conv0<<<nb((long)B_ * 32), 256, 0, stream>>>(
      feat_erb, erb0_buf, P(I_ec0 + 1), P(I_ec0 + 2), P(I_ec0 + 3), P(I_ec0 + 0),
      e0, out + O_erbbuf);
  k_dw_freq3<<<nb((long)B_ * 16 * 64), 256, 0, stream>>>(e0, nullptr, P(I_ec1 + 1), abuf, 32, 16, 2);
  gemm(abuf, 64, 0, Pw_e1, 0, P(I_ec1 + 0), 0, e1, 64, 0, B_ * 16, 64, 64, 1, 1);
  k_dw_freq3<<<nb((long)B_ * 8 * 64), 256, 0, stream>>>(e1, nullptr, P(I_ec2 + 1), abuf, 16, 8, 2);
  gemm(abuf, 64, 0, Pw_e2, 0, P(I_ec2 + 0), 0, e2, 64, 0, B_ * 8, 64, 64, 1, 1);
  k_dw_freq3<<<nb((long)B_ * 8 * 64), 256, 0, stream>>>(e2, nullptr, P(I_ec3 + 1), abuf, 8, 8, 1);
  gemm(abuf, 64, 0, Pw_e3, 0, P(I_ec3 + 0), 0, e3, 64, 0, B_ * 8, 64, 64, 1, 1);
  k_df_conv0<<<nb((long)B_ * 96), 256, 0, stream>>>(
      feat_spec, df0_buf, P(I_dfc0 + 1), P(I_dfc0 + 2), P(I_dfc0 + 3), P(I_dfc0 + 0),
      c0, out + O_dfbuf);
  k_dw_freq3<<<nb((long)B_ * 48 * 64), 256, 0, stream>>>(c0, nullptr, P(I_dfc1 + 1), abuf, 96, 48, 2);
  gemm(abuf, 64, 0, Pw_d1, 0, P(I_dfc1 + 0), 0, c1, 64, 0, B_ * 48, 64, 64, 1, 1);
  gemm(c1, 3072, 384, Gw_fce, 384 * 64, nullptr, 0, cemb, 512, 64, B_, 64, 384, 1, 8);
  k_add<<<nb((long)B_ * 512), 256, 0, stream>>>(e3, cemb, emb, (long)B_ * 512);
  gemm(emb, 512, 64, Gw_enc, 64 * 64, nullptr, 0, xin, 512, 64, B_, 64, 64, 1, 8);
  gemm(xin, 512, 0, Wx_enc, 0, P(I_enc_bx), 0, gx, 1536, 0, B_, 1536, 512, 0, 1);
  gemm(enc_state, 512, 0, Wh_enc, 0, P(I_enc_bh), 0, gh, 1536, 0, B_, 1536, 512, 0, 1);
  k_gru_gate<<<nb((long)B_ * 512), 256, 0, stream>>>(
      gx, gh, enc_state, 512, 0, embO, 512, 0, out + O_enc, 512, 0, B_, 512);
  k_dense1<<<nb(B_), 256, 0, stream>>>(embO, P(I_lsnr_W), P(I_lsnr_b),
                                       out + O_lsnr, 512, 50.f, -15.f);

  // ================= Phase 2: ERB decoder =================
  gemm(embO, 512, 64, Gw_ein, 64 * 64, nullptr, 0, xin, 512, 64, B_, 64, 64, 1, 8);
  gemm(xin, 512, 0, Wx_erb, 0, P(I_erb_bx), 0, gx, 1536, 0, B_, 1536, 512, 0, 1);
  gemm(erb_state, 512, 0, Wh_erb, 0, P(I_erb_bh), 0, gh, 1536, 0, B_, 1536, 512, 0, 1);
  k_gru_gate<<<nb((long)B_ * 512), 256, 0, stream>>>(
      gx, gh, erb_state, 512, 0, ta, 512, 0, out + O_erbst, 512, 0, B_, 512);
  k_add<<<nb((long)B_ * 512), 256, 0, stream>>>(ta, xin, tb, (long)B_ * 512);
  gemm(tb, 512, 64, Gw_eout, 64 * 64, nullptr, 0, tc, 512, 64, B_, 64, 64, 1, 8);
  gemm(e3, 64, 0, Pw_c3p, 0, P(I_c3p + 0), 0, px, 64, 0, B_ * 8, 64, 64, 1, 1);
  k_dw_freq3<<<nb((long)B_ * 8 * 64), 256, 0, stream>>>(px, tc, P(I_ct3 + 1), abuf, 8, 8, 1);
  gemm(abuf, 64, 0, Pw_t3, 0, P(I_ct3 + 0), 0, xx, 64, 0, B_ * 8, 64, 64, 1, 1);
  gemm(e2, 64, 0, Pw_c2p, 0, P(I_c2p + 0), 0, px, 64, 0, B_ * 8, 64, 64, 1, 1);
  k_dwt_freq3<<<nb((long)B_ * 16 * 64), 256, 0, stream>>>(px, xx, P(I_ct2 + 1), abuf, 8);
  gemm(abuf, 64, 0, Pw_t2, 0, P(I_ct2 + 0), 0, xx, 64, 0, B_ * 16, 64, 64, 1, 1);
  gemm(e1, 64, 0, Pw_c1p, 0, P(I_c1p + 0), 0, px, 64, 0, B_ * 16, 64, 64, 1, 1);
  k_dwt_freq3<<<nb((long)B_ * 32 * 64), 256, 0, stream>>>(px, xx, P(I_ct1 + 1), abuf, 16);
  gemm(abuf, 64, 0, Pw_t1, 0, P(I_ct1 + 0), 0, xx, 64, 0, B_ * 32, 64, 64, 1, 1);
  gemm(e0, 64, 0, Pw_c0p, 0, P(I_c0p + 0), 0, px, 64, 0, B_ * 32, 64, 64, 1, 1);
  k_conv0_out<<<nb((long)B_ * 32), 256, 0, stream>>>(
      px, xx, P(I_c0out + 1), P(I_c0out + 2), P(I_c0out + 3), P(I_c0out + 0), out + O_m);

  // ================= Phase 3: DF decoder =================
  gemm(embO, 512, 64, Gw_din, 64 * 64, nullptr, 0, xin, 512, 64, B_, 64, 64, 1, 8);
  gemm(xin, 512, 0, Wx_df0, 0, P(I_df0_bx), 0, gx, 1536, 0, B_, 1536, 512, 0, 1);
  gemm(df_state, 1024, 0, Wh_df0, 0, P(I_df0_bh), 0, gh, 1536, 0, B_, 1536, 512, 0, 1);
  k_gru_gate<<<nb((long)B_ * 512), 256, 0, stream>>>(
      gx, gh, df_state, 1024, 0, ta, 512, 0, out + O_dfst, 1024, 0, B_, 512);
  gemm(ta, 512, 0, Wx_df1, 0, P(I_df1_bx), 0, gx, 1536, 0, B_, 1536, 512, 0, 1);
  gemm(df_state + 512, 1024, 0, Wh_df1, 0, P(I_df1_bh), 0, gh, 1536, 0, B_, 1536, 512, 0, 1);
  k_gru_gate<<<nb((long)B_ * 512), 256, 0, stream>>>(
      gx, gh, df_state, 1024, 512, tb, 512, 0, out + O_dfst, 1024, 512, B_, 512);
  k_add<<<nb((long)B_ * 512), 256, 0, stream>>>(tb, xin, td, (long)B_ * 512);
  k_dense1<<<nb(B_), 256, 0, stream>>>(td, P(I_dfa_W), P(I_dfa_b),
                                       out + O_alpha, 512, 1.f, 0.f);
  k_dw_time5<<<nb((long)B_ * 96 * 64), 256, 0, stream>>>(
      c0, convp_buf, P(I_dfcp + 1), abuf, out + O_cpbuf);
  gemm(abuf, 64, 0, Pw_cp, 0, P(I_dfcp + 0), 0, c0f, 10, 0, B_ * 96, 10, 64, 1, 1);
  gemm(td, 512, 64, Gw_dout, 120 * 64, nullptr, 0, gout, 960, 120, B_, 120, 64, 3, 8);
  k_coefs<<<nb((long)B_ * 96 * 10), 256, 0, stream>>>(gout, c0f, out + O_coefs);
}
